// PhysBiformerBlock_32564442038683
// MI455X (gfx1250) — matrix operand order repaired and checked by reading, not compiled, not run on hardware
//
#include <hip/hip_runtime.h>
#include <hip/hip_bf16.h>
#include <math.h>

// ---------------------------------------------------------------------------
// Types / constants
// ---------------------------------------------------------------------------
typedef __bf16 v16bf __attribute__((ext_vector_type(16)));
typedef float  v8f   __attribute__((ext_vector_type(8)));
typedef unsigned v4u __attribute__((ext_vector_type(4)));
typedef unsigned v8u __attribute__((ext_vector_type(8)));

#define C_DIM   256
#define LSP     8192        // 8*32*32 spatial sites
#define TT      2
#define M_TOK   16384       // TT * LSP tokens
#define R_REG   32
#define P_REG   256
#define NHEAD   8
#define DHEAD   32
#define TOPK_   4

// ---------------------------------------------------------------------------
// bf16 helpers (storage-only; no __bf16 arithmetic)
// ---------------------------------------------------------------------------
__device__ __forceinline__ unsigned short f2bf(float f) {
    unsigned int u = __float_as_uint(f);
    unsigned int r = u + 0x7FFFu + ((u >> 16) & 1u);
    return (unsigned short)(r >> 16);
}
__device__ __forceinline__ float bf2f(unsigned short h) {
    return __uint_as_float(((unsigned int)h) << 16);
}

// generic pointer (to __shared__) -> 32-bit LDS byte address
__device__ __forceinline__ unsigned lds_addr_of(const void* p) {
    return (unsigned)(unsigned long long)
        (const __attribute__((address_space(3))) void*)p;
}

// A-fragment (16x32 bf16, MxK): lane<16 -> K 0..7,16..23 ; lane>=16 -> K 8..15,24..31
__device__ __forceinline__ v16bf load_fragA(const unsigned short* row, int khalf) {
    union { v16bf v; uint4 q[2]; } u;
    u.q[0] = *(const uint4*)(row + khalf * 8);
    u.q[1] = *(const uint4*)(row + 16 + khalf * 8);
    return u.v;
}
// B-fragment (32x16 bf16, KxN), row = column data (32 K-contiguous elems):
// lane<16 -> K 0..7,16..23 ; lane>=16 -> K 8..15,24..31
__device__ __forceinline__ v16bf load_fragB(const unsigned short* col, int khalf) {
    union { v16bf v; uint4 q[2]; } u;
    u.q[0] = *(const uint4*)(col + khalf * 8);
    u.q[1] = *(const uint4*)(col + 16 + khalf * 8);
    return u.v;
}

// ---------------------------------------------------------------------------
// TDM: 2D tile (128 rows x 32 bf16) from global [.., Kdim] into LDS.
// D# group0 (128b): count=1 | lds_addr | global_addr | type=2
// D# group1 (256b): data_size=2B, tensor_dim0=Kdim, tensor_dim1=Mdim,
//                   tile_dim0=32, tile_dim1=128, stride0=Kdim
// Issued by one wave; completion via TENSORcnt.
// ---------------------------------------------------------------------------
__device__ __forceinline__ void tdm_load_a_tile(const unsigned short* gsrc,
                                                unsigned lds_off,
                                                int Kdim, int Mdim) {
    unsigned long long ga = (unsigned long long)gsrc;
    v4u g0;
    g0[0] = 1u;                                               // count = 1 (valid)
    g0[1] = lds_off;                                          // lds_addr
    g0[2] = (unsigned)(ga & 0xffffffffu);                     // global_addr[31:0]
    g0[3] = (unsigned)((ga >> 32) & 0x01ffffffu) | (2u << 30);// addr[56:32] | type=2
    v8u g1;
    g1[0] = 1u << 16;                                         // data_size = 1 (2 bytes)
    g1[1] = ((unsigned)Kdim & 0xffffu) << 16;                 // tensor_dim0[15:0]
    g1[2] = (((unsigned)Kdim >> 16) & 0xffffu)
          | (((unsigned)Mdim & 0xffffu) << 16);               // dim0[31:16] | dim1[15:0]
    g1[3] = (((unsigned)Mdim >> 16) & 0xffffu) | (32u << 16); // dim1[31:16] | tile_dim0=32
    g1[4] = 128u;                                             // tile_dim1=128, tile_dim2=0
    g1[5] = (unsigned)Kdim;                                   // tensor_dim0_stride[31:0]
    g1[6] = 0u;
    g1[7] = 0u;
    asm volatile("tensor_load_to_lds %0, %1" :: "s"(g0), "s"(g1) : "memory");
}

// per-lane async global->LDS copy of 8 bytes (ASYNCcnt tracked)
__device__ __forceinline__ void async_ld_b64(unsigned lds_off, const void* gaddr) {
    asm volatile("global_load_async_to_lds_b64 %0, %1, off"
                 :: "v"(lds_off), "v"((unsigned long long)gaddr) : "memory");
}
__device__ __forceinline__ void wait_async0() {
    asm volatile("s_wait_asynccnt 0x0" ::: "memory");
}

// spatial site s (lt*1024 + lh*32 + lw) + time t  ->  region-major token id
__device__ __forceinline__ int tok_of(int t, int s) {
    int lt = s >> 10, lh = (s >> 5) & 31, lw = s & 31;
    int r = ((lt >> 2) << 4) + ((lh >> 3) << 2) + (lw >> 3);
    int p = ((lt & 3) << 6) + ((lh & 7) << 3) + (lw & 7);
    return ((t << 5) + r) * P_REG + p;
}

// ---------------------------------------------------------------------------
// BatchNorm stats: fold into per-channel (scale, shift)
// ---------------------------------------------------------------------------
__global__ __launch_bounds__(256)
void bn_stats(const float* __restrict__ x, const float* __restrict__ g,
              const float* __restrict__ b, float* __restrict__ stats) {
    int c = blockIdx.x, tid = threadIdx.x;
    float s = 0.f, s2 = 0.f;
    for (int t = 0; t < TT; ++t) {
        const float* p = x + ((size_t)t * C_DIM + c) * LSP;
        for (int i = tid; i < LSP; i += 256) { float v = p[i]; s += v; s2 += v * v; }
    }
    __shared__ float rs[256], rs2[256];
    rs[tid] = s; rs2[tid] = s2; __syncthreads();
    for (int st = 128; st > 0; st >>= 1) {
        if (tid < st) { rs[tid] += rs[tid + st]; rs2[tid] += rs2[tid + st]; }
        __syncthreads();
    }
    if (tid == 0) {
        float inv  = 1.f / (float)(TT * LSP);
        float mean = rs[0] * inv;
        float var  = rs2[0] * inv - mean * mean;
        float sc   = g[c] * rsqrtf(var + 1e-5f);
        stats[c * 2 + 0] = sc;
        stats[c * 2 + 1] = b[c] - mean * sc;
    }
}

// ---------------------------------------------------------------------------
// BN apply + 2-step LIF (tau=2, vth=1, soft reset) -> bf16 spikes, region-major
// ---------------------------------------------------------------------------
__global__ __launch_bounds__(256)
void bn_lif(const float* __restrict__ x, const float* __restrict__ stats,
            unsigned short* __restrict__ spk) {
    int c = threadIdx.x, s = blockIdx.x;
    float sc = stats[c * 2], sh = stats[c * 2 + 1];
    float x0 = x[((size_t)0 * C_DIM + c) * LSP + s] * sc + sh;
    float x1 = x[((size_t)1 * C_DIM + c) * LSP + s] * sc + sh;
    float v  = 0.5f * x0;                       // v += (x - v)/tau, v0=0
    float sp0 = (v - 1.0f >= 0.f) ? 1.f : 0.f;
    v -= sp0;
    v = 0.5f * (v + x1);
    float sp1 = (v - 1.0f >= 0.f) ? 1.f : 0.f;
    spk[(size_t)tok_of(0, s) * C_DIM + c] = f2bf(sp0);
    spk[(size_t)tok_of(1, s) * C_DIM + c] = f2bf(sp1);
}

// ---------------------------------------------------------------------------
// Tiled WMMA GEMM:  Out[M,N] = act(A[M,K](bf16) @ W[K,N](f32->bf16) + bias)
// Block tile 128x128, K-step 32, 8 waves in 2(M)x4(N); wave tile 64x32.
// A tiles streamed by the Tensor Data Mover (double-buffered, TENSORcnt);
// W tiles converted f32->bf16 by the VALU in parallel with the DMA.
// MODE 0: f32 out ; 1: bf16 out ; 2: bf16 out + exact GELU
// ---------------------------------------------------------------------------
template<int MODE>
__global__ __launch_bounds__(256)
void gemm_bf16(const unsigned short* __restrict__ A, const float* __restrict__ W,
               const float* __restrict__ bias, void* __restrict__ Out,
               int Mdim, int Ndim, int Kdim) {
    __shared__ __align__(16) unsigned short As[2][128][32];  // [buf][m][k]
    __shared__ __align__(16) unsigned short Bs[128][32];     // [n][k] (transposed)
    const int tid = threadIdx.x, lane = tid & 31, wid = tid >> 5;
    const int wm = wid & 1, wn = wid >> 1;
    const int bm = blockIdx.y * 128, bn = blockIdx.x * 128;
    const int nk = Kdim >> 5;

    v8f vzero;
    #pragma unroll
    for (int i = 0; i < 8; ++i) vzero[i] = 0.f;
    v8f acc[4][2];
    #pragma unroll
    for (int i = 0; i < 4; ++i) { acc[i][0] = vzero; acc[i][1] = vzero; }

    const unsigned asb[2] = { lds_addr_of(&As[0][0][0]), lds_addr_of(&As[1][0][0]) };

    // prime the TDM pipe with tile 0
    if (wid == 0)
        tdm_load_a_tile(A + (size_t)bm * Kdim, asb[0], Kdim, Mdim);

    for (int it = 0; it < nk; ++it) {
        const int k0 = it << 5;
        __syncthreads();                       // prior consumers of As[(it+1)&1] / Bs done
        if (wid == 0 && it + 1 < nk)
            tdm_load_a_tile(A + (size_t)bm * Kdim + (k0 + 32),
                            asb[(it + 1) & 1], Kdim, Mdim);
        // --- stage W tile 32x128 f32 -> bf16, transposed into Bs[n][k] ---
        {
            int n = tid >> 1, kc = (tid & 1) * 16;
            unsigned short tmp[16] __attribute__((aligned(16)));
            #pragma unroll
            for (int i = 0; i < 16; ++i)
                tmp[i] = f2bf(W[(size_t)(k0 + kc + i) * Ndim + (bn + n)]);
            *(uint4*)(&Bs[n][kc])     = *(const uint4*)(&tmp[0]);
            *(uint4*)(&Bs[n][kc + 8]) = *(const uint4*)(&tmp[8]);
            if (k0 + 32 < Kdim)
                __builtin_prefetch(&W[(size_t)(k0 + 32 + kc) * Ndim + (bn + n)], 0, 1);
        }
        if (wid == 0) {                        // tile `it` landed in LDS?
            if (it + 1 < nk) __builtin_amdgcn_s_wait_tensorcnt(1);
            else             __builtin_amdgcn_s_wait_tensorcnt(0);
        }
        __syncthreads();
        // --- fragments + 8 WMMAs ---
        const unsigned short (*Ac)[32] = As[it & 1];
        v16bf af[4], bfv[2];
        #pragma unroll
        for (int mt = 0; mt < 4; ++mt)
            af[mt] = load_fragA(&Ac[wm * 64 + mt * 16 + (lane & 15)][0], lane >> 4);
        #pragma unroll
        for (int nt = 0; nt < 2; ++nt)
            bfv[nt] = load_fragB(&Bs[wn * 32 + nt * 16 + (lane & 15)][0], lane >> 4);
        #pragma unroll
        for (int mt = 0; mt < 4; ++mt)
            #pragma unroll
            for (int nt = 0; nt < 2; ++nt)
                acc[mt][nt] = __builtin_amdgcn_wmma_f32_16x16x32_bf16(
                    false, af[mt], false, bfv[nt], (short)0, acc[mt][nt], false, false);
    }

    // --- epilogue ---
    #pragma unroll
    for (int mt = 0; mt < 4; ++mt) {
        #pragma unroll
        for (int nt = 0; nt < 2; ++nt) {
            int col = bn + wn * 32 + nt * 16 + (lane & 15);
            float bb = bias[col];
            int rowBase = bm + wm * 64 + mt * 16 + ((lane >> 4) << 3);
            #pragma unroll
            for (int j = 0; j < 8; ++j) {
                float v = acc[mt][nt][j] + bb;
                if (MODE == 2) v = 0.5f * v * (1.f + erff(v * 0.70710678118654752f));
                size_t o = (size_t)(rowBase + j) * Ndim + col;
                if (MODE == 0) ((float*)Out)[o] = v;
                else           ((unsigned short*)Out)[o] = f2bf(v);
            }
        }
    }
}

// ---------------------------------------------------------------------------
// Region means of q and k:  qm/km [N*R, C]
// ---------------------------------------------------------------------------
__global__ __launch_bounds__(256)
void region_mean(const unsigned short* __restrict__ qkv,
                 float* __restrict__ qm, float* __restrict__ km) {
    int c = threadIdx.x, r = blockIdx.x, n = blockIdx.y, sel = blockIdx.z;
    size_t base = (size_t)(n * R_REG + r) * P_REG * 768 + sel * C_DIM + c;
    float sum = 0.f;
    for (int p = 0; p < P_REG; ++p) sum += bf2f(qkv[base + (size_t)p * 768]);
    float* dst = sel ? km : qm;
    dst[(n * R_REG + r) * C_DIM + c] = sum * (1.f / (float)P_REG);
}

// ---------------------------------------------------------------------------
// Coarse routing: a_r = qm@km^T per (n,r); top-4 (desc, lower idx on ties)
// ---------------------------------------------------------------------------
__global__ __launch_bounds__(32)
void route_topk(const float* __restrict__ qm, const float* __restrict__ km,
                int* __restrict__ idxb) {
    int nr = blockIdx.x, n = nr >> 5, s = threadIdx.x;
    __shared__ float a[R_REG];
    const float* q = qm + (size_t)nr * C_DIM;
    const float* k = km + (size_t)(n * R_REG + s) * C_DIM;
    float dot = 0.f;
    for (int c = 0; c < C_DIM; ++c) dot += q[c] * k[c];
    a[s] = dot;
    __syncthreads();
    if (s == 0) {
        unsigned usedMask = 0;
        for (int t = 0; t < TOPK_; ++t) {
            int best = 0; float bv = -3.4e38f;
            for (int i = 0; i < R_REG; ++i)
                if (!((usedMask >> i) & 1u) && a[i] > bv) { bv = a[i]; best = i; }
            usedMask |= 1u << best;
            idxb[nr * TOPK_ + t] = best;
        }
    }
}

// ---------------------------------------------------------------------------
// Routed attention, flash-style. Block = (head, region, n). 8 waves x 32 q-rows.
// Gathered K tiles streamed by per-lane async global->LDS (ASYNCcnt);
// V transposed through VGPRs; S & PV both via bf16 WMMA.
// ---------------------------------------------------------------------------
__global__ __launch_bounds__(256)
void bra_attn(const unsigned short* __restrict__ qkv, const int* __restrict__ idxb,
              unsigned short* __restrict__ attnout) {
    const int h = blockIdx.x, r = blockIdx.y, n = blockIdx.z;
    const int tid = threadIdx.x, lane = tid & 31, wid = tid >> 5;
    __shared__ __align__(16) unsigned short Ksh[32][32];      // [key][d]
    __shared__ __align__(16) unsigned short Vsh[32][32];      // [d][key]  (V^T)
    __shared__ __align__(16) unsigned short Psh[8][32][32];   // per-wave P tiles

    v8f vzero;
    #pragma unroll
    for (int i = 0; i < 8; ++i) vzero[i] = 0.f;

    // q fragments straight from global (region-major layout => contiguous)
    const int p0 = wid * 32;
    const size_t qbase = (size_t)(n * R_REG + r) * P_REG * 768 + h * DHEAD;
    v16bf aq[2];
    #pragma unroll
    for (int mt = 0; mt < 2; ++mt)
        aq[mt] = load_fragA(qkv + qbase + (size_t)(p0 + mt * 16 + (lane & 15)) * 768,
                            lane >> 4);

    float m_run[2][8], l_run[2][8];
    v8f o_[2][2];
    #pragma unroll
    for (int mt = 0; mt < 2; ++mt) {
        o_[mt][0] = vzero; o_[mt][1] = vzero;
        #pragma unroll
        for (int j = 0; j < 8; ++j) { m_run[mt][j] = -3.0e38f; l_run[mt][j] = 0.f; }
    }

    const int rbase = (n * R_REG + r) * TOPK_;
    const float sm_scale = 0.17677669529663687f;   // Dh^-0.5

    for (int kt = 0; kt < (TOPK_ * P_REG) / 32; ++kt) {
        __syncthreads();
        // --- cooperative gather of 32 keys ---
        {
            int key = tid >> 3, dp = (tid & 7) * 4;
            int kk  = kt * 32 + key;
            int reg = idxb[rbase + (kk >> 8)];
            size_t krow = ((size_t)(n * R_REG + reg) * P_REG + (kk & 255)) * 768;
            // K row-major: per-lane async copy straight into LDS
            async_ld_b64(lds_addr_of(&Ksh[key][dp]),
                         qkv + krow + C_DIM + h * DHEAD + dp);
            // V transposed through VGPRs
            const unsigned short* vp = qkv + krow + 2 * C_DIM + h * DHEAD + dp;
            #pragma unroll
            for (int i = 0; i < 4; ++i) Vsh[dp + i][key] = vp[i];
        }
        wait_async0();
        __syncthreads();

        // --- S = q . k^T (4 WMMAs) ---
        v16bf bk[2];
        #pragma unroll
        for (int nt = 0; nt < 2; ++nt)
            bk[nt] = load_fragB(&Ksh[nt * 16 + (lane & 15)][0], lane >> 4);
        v8f S[2][2];
        #pragma unroll
        for (int mt = 0; mt < 2; ++mt)
            #pragma unroll
            for (int nt = 0; nt < 2; ++nt)
                S[mt][nt] = __builtin_amdgcn_wmma_f32_16x16x32_bf16(
                    false, aq[mt], false, bk[nt], (short)0, vzero, false, false);

        // --- online softmax (row reductions via half-wave shuffles) ---
        #pragma unroll
        for (int mt = 0; mt < 2; ++mt) {
            #pragma unroll
            for (int j = 0; j < 8; ++j) {
                float s0 = S[mt][0][j] * sm_scale;
                float s1 = S[mt][1][j] * sm_scale;
                float mx = fmaxf(s0, s1);
                mx = fmaxf(mx, __shfl_xor(mx, 1, 32));
                mx = fmaxf(mx, __shfl_xor(mx, 2, 32));
                mx = fmaxf(mx, __shfl_xor(mx, 4, 32));
                mx = fmaxf(mx, __shfl_xor(mx, 8, 32));
                float mnew = fmaxf(m_run[mt][j], mx);
                float corr = __expf(m_run[mt][j] - mnew);
                float e0 = __expf(s0 - mnew), e1 = __expf(s1 - mnew);
                float rs = e0 + e1;
                rs += __shfl_xor(rs, 1, 32);
                rs += __shfl_xor(rs, 2, 32);
                rs += __shfl_xor(rs, 4, 32);
                rs += __shfl_xor(rs, 8, 32);
                l_run[mt][j] = l_run[mt][j] * corr + rs;
                m_run[mt][j] = mnew;
                o_[mt][0][j] *= corr;
                o_[mt][1][j] *= corr;
                int lr = mt * 16 + j + ((lane >> 4) << 3);
                Psh[wid][lr][(lane & 15)]      = f2bf(e0);
                Psh[wid][lr][16 + (lane & 15)] = f2bf(e1);
            }
        }
        __syncthreads();

        // --- O += P @ V (4 WMMAs) ---
        v16bf ap[2], bv[2];
        #pragma unroll
        for (int mt = 0; mt < 2; ++mt)
            ap[mt] = load_fragA(&Psh[wid][mt * 16 + (lane & 15)][0], lane >> 4);
        #pragma unroll
        for (int dt = 0; dt < 2; ++dt)
            bv[dt] = load_fragB(&Vsh[dt * 16 + (lane & 15)][0], lane >> 4);
        #pragma unroll
        for (int mt = 0; mt < 2; ++mt)
            #pragma unroll
            for (int dt = 0; dt < 2; ++dt)
                o_[mt][dt] = __builtin_amdgcn_wmma_f32_16x16x32_bf16(
                    false, ap[mt], false, bv[dt], (short)0, o_[mt][dt], false, false);
    }

    // --- normalize + store bf16 into attnout[token, C] ---
    #pragma unroll
    for (int mt = 0; mt < 2; ++mt) {
        #pragma unroll
        for (int dt = 0; dt < 2; ++dt) {
            int col = h * DHEAD + dt * 16 + (lane & 15);
            #pragma unroll
            for (int j = 0; j < 8; ++j) {
                int prow = p0 + mt * 16 + j + ((lane >> 4) << 3);
                float val = o_[mt][dt][j] / l_run[mt][j];
                attnout[((size_t)(n * R_REG + r) * P_REG + prow) * C_DIM + col] = f2bf(val);
            }
        }
    }
}

// ---------------------------------------------------------------------------
// LEPE: depthwise 3x3x3 conv on v, added into bf16 attention output
// ---------------------------------------------------------------------------
__global__ __launch_bounds__(256)
void lepe_add(const unsigned short* __restrict__ qkv, const float* __restrict__ w,
              unsigned short* __restrict__ attnout) {
    int i = blockIdx.x * 256 + threadIdx.x;        // over T*C*LSP
    int t = i >> 21, c = (i >> 13) & 255, s = i & 8191;
    int lt = s >> 10, lh = (s >> 5) & 31, lw = s & 31;
    const float* wc = w + c * 27;
    float acc = 0.f;
    #pragma unroll
    for (int dz = -1; dz <= 1; ++dz) {
        int z = lt + dz; if (z < 0 || z >= 8) continue;
        #pragma unroll
        for (int dy = -1; dy <= 1; ++dy) {
            int y = lh + dy; if (y < 0 || y >= 32) continue;
            #pragma unroll
            for (int dx = -1; dx <= 1; ++dx) {
                int xw = lw + dx; if (xw < 0 || xw >= 32) continue;
                int ns = (z << 10) + (y << 5) + xw;
                acc += bf2f(qkv[(size_t)tok_of(t, ns) * 768 + 2 * C_DIM + c]) *
                       wc[(dz + 1) * 9 + (dy + 1) * 3 + (dx + 1)];
            }
        }
    }
    size_t o = (size_t)tok_of(t, s) * C_DIM + c;
    attnout[o] = f2bf(bf2f(attnout[o]) + acc);
}

// ---------------------------------------------------------------------------
// Residual: out[chan-major] = base + scale * branch[token-major]
// ---------------------------------------------------------------------------
template<bool ACC>
__global__ __launch_bounds__(256)
void residual(const float* __restrict__ xin, const float* __restrict__ gout,
              const float* __restrict__ scale, float* __restrict__ out) {
    int i = blockIdx.x * 256 + threadIdx.x;
    int t = i >> 21, c = (i >> 13) & 255, s = i & 8191;
    float add = scale[0] * gout[(size_t)tok_of(t, s) * C_DIM + c];
    out[i] = (ACC ? out[i] : xin[i]) + add;
}

// ---------------------------------------------------------------------------
// Launch
// ---------------------------------------------------------------------------
extern "C" void kernel_launch(void* const* d_in, const int* in_sizes, int n_in,
                              void* d_out, int out_size, void* d_ws, size_t ws_size,
                              hipStream_t stream) {
    const float* x      = (const float*)d_in[0];
    const float* bn1_g  = (const float*)d_in[1];
    const float* bn1_b  = (const float*)d_in[2];
    const float* wqkv   = (const float*)d_in[3];
    const float* bqkv   = (const float*)d_in[4];
    const float* lepe_w = (const float*)d_in[5];
    const float* wo     = (const float*)d_in[6];
    const float* bo     = (const float*)d_in[7];
    const float* bn2_g  = (const float*)d_in[8];
    const float* bn2_b  = (const float*)d_in[9];
    const float* ffn_w1 = (const float*)d_in[10];
    const float* ffn_b1 = (const float*)d_in[11];
    const float* ffn_w2 = (const float*)d_in[12];
    const float* ffn_b2 = (const float*)d_in[13];
    const float* scale  = (const float*)d_in[14];
    float* out = (float*)d_out;

    char* ws = (char*)d_ws; size_t off = 0;
    auto alloc = [&](size_t bytes) -> void* {
        void* p = ws + off; off += (bytes + 255) & ~(size_t)255; return p;
    };
    float*          stats1 = (float*)alloc(C_DIM * 2 * sizeof(float));
    float*          stats2 = (float*)alloc(C_DIM * 2 * sizeof(float));
    unsigned short* spk    = (unsigned short*)alloc((size_t)M_TOK * C_DIM * 2);       // 8 MB (s1 then s2)
    unsigned short* qkvb   = (unsigned short*)alloc((size_t)M_TOK * 3 * C_DIM * 2);   // 24 MB
    float*          qm     = (float*)alloc((size_t)TT * R_REG * C_DIM * 4);
    float*          km     = (float*)alloc((size_t)TT * R_REG * C_DIM * 4);
    int*            idxb   = (int*)alloc((size_t)TT * R_REG * TOPK_ * 4);
    unsigned short* attno  = (unsigned short*)alloc((size_t)M_TOK * C_DIM * 2);       // 8 MB
    float*          gout   = (float*)alloc((size_t)M_TOK * C_DIM * 4);                // 16 MB (a_out then f)
    unsigned short* hbuf   = (unsigned short*)alloc((size_t)M_TOK * 4 * C_DIM * 2);   // 32 MB

    const int nElem = TT * C_DIM * LSP;   // 4,194,304

    // --- attention branch ---
    bn_stats<<<C_DIM, 256, 0, stream>>>(x, bn1_g, bn1_b, stats1);
    bn_lif  <<<LSP,   C_DIM, 0, stream>>>(x, stats1, spk);
    gemm_bf16<1><<<dim3(768 / 128, M_TOK / 128), 256, 0, stream>>>(
        spk, wqkv, bqkv, qkvb, M_TOK, 768, C_DIM);
    region_mean<<<dim3(R_REG, TT, 2), C_DIM, 0, stream>>>(qkvb, qm, km);
    route_topk <<<TT * R_REG, 32, 0, stream>>>(qm, km, idxb);
    bra_attn   <<<dim3(NHEAD, R_REG, TT), 256, 0, stream>>>(qkvb, idxb, attno);
    lepe_add   <<<nElem / 256, 256, 0, stream>>>(qkvb, lepe_w, attno);
    gemm_bf16<0><<<dim3(256 / 128, M_TOK / 128), 256, 0, stream>>>(
        attno, wo, bo, gout, M_TOK, C_DIM, C_DIM);
    residual<false><<<nElem / 256, 256, 0, stream>>>(x, gout, scale, out);

    // --- FFN branch ---
    bn_stats<<<C_DIM, 256, 0, stream>>>(out, bn2_g, bn2_b, stats2);
    bn_lif  <<<LSP,   C_DIM, 0, stream>>>(out, stats2, spk);
    gemm_bf16<2><<<dim3(1024 / 128, M_TOK / 128), 256, 0, stream>>>(
        spk, ffn_w1, ffn_b1, hbuf, M_TOK, 1024, C_DIM);
    gemm_bf16<0><<<dim3(256 / 128, M_TOK / 128), 256, 0, stream>>>(
        hbuf, ffn_w2, ffn_b2, gout, M_TOK, C_DIM, 1024);
    residual<true><<<nElem / 256, 256, 0, stream>>>(out, gout, scale, out);
}
